// EGCL_68556267979332
// MI455X (gfx1250) — matrix operand order repaired and checked by reading, not compiled, not run on hardware
//
#include <hip/hip_runtime.h>
#include <hip/hip_bf16.h>

typedef _Float16 f16;
typedef __attribute__((ext_vector_type(16))) _Float16 v16h;
typedef __attribute__((ext_vector_type(8)))  _Float16 v8h;
typedef __attribute__((ext_vector_type(8)))  float    v8f;
typedef __attribute__((ext_vector_type(4)))  unsigned int u32x4;
typedef __attribute__((ext_vector_type(8)))  int      i32x8;

#define STRD  136   // f16 row stride for K=128 tiles (272B: 16B-aligned, bank-staggered)
#define STRD3 264   // f16 row stride for K=256 rows (528B, 16B-aligned)

__device__ __forceinline__ float siluf(float x) { return x / (1.0f + __expf(-x)); }
__device__ __forceinline__ float sigmf_(float x) { return 1.0f / (1.0f + __expf(-x)); }

// A-matrix 16x32 f16 fragment (ISA 7.12.2): lane L<16 -> row L, K = kb+{0..7,16..23};
// lane L+16 -> row L, K = kb+{8..15,24..31}.
__device__ __forceinline__ v16h frag_ld_a(const f16* __restrict__ buf, int row0, int kb, int stride) {
  const int lane = threadIdx.x & 31;
  const int r = row0 + (lane & 15);
  const int k = kb + ((lane >> 4) << 3);
  const f16* p = buf + r * stride + k;
  v8h lo = *(const v8h*)(p);
  v8h hi = *(const v8h*)(p + 16);
  return __builtin_shufflevector(lo, hi, 0,1,2,3,4,5,6,7,8,9,10,11,12,13,14,15);
}

// B-matrix 32x16 f16 fragment (matches A layout): lane L<16 -> col L,
// K = kb+{0..7,16..23}; lane L+16 -> col L, K = kb+{8..15,24..31}. `buf` holds B^T rows [n][k].
__device__ __forceinline__ v16h frag_ld_b(const f16* __restrict__ buf, int col0, int kb, int stride) {
  const int lane = threadIdx.x & 31;
  const int n = col0 + (lane & 15);
  const int k = kb + ((lane >> 4) << 3);
  const f16* p = buf + n * stride + k;
  v8h lo = *(const v8h*)(p);
  v8h hi = *(const v8h*)(p + 16);
  return __builtin_shufflevector(lo, hi, 0,1,2,3,4,5,6,7,8,9,10,11,12,13,14,15);
}

// TDM: async 2-D tile DMA global->LDS (CDNA5 ISA-8). 2 descriptor groups (<=2D tensor).
// Optional LDS padding: after (1<<(pad_interval+1)) DWORDs, skip (pad_amount+1) DWORDs.
__device__ __forceinline__ void tdm_load_2d(const void* gptr, void* lptr,
                                            int tile_w, int tile_h, int gstride_elems,
                                            int pad_interval, int pad_amount, int pad_en) {
  unsigned long long ga = (unsigned long long)gptr;
  unsigned la = (unsigned)(unsigned long long)lptr;   // ISA 10.2: LDS flat addr low 32 bits = LDS byte address
  u32x4 g0;
  g0[0] = 1u;                                         // count=1 valid descriptor
  g0[1] = la;                                         // lds_addr
  g0[2] = (unsigned)ga;                               // global_addr[31:0]
  g0[3] = (unsigned)(ga >> 32) | (2u << 30);          // global_addr[56:32] | type=2
  i32x8 g1;
  g1[0] = (2 << 16) | (pad_en << 20) | (pad_interval << 22) | (pad_amount << 25); // data_size=4B
  g1[1] = (tile_w & 0xFFFF) << 16;                    // tensor_dim0[15:0]
  g1[2] = ((unsigned)tile_w >> 16) | (tile_h << 16);  // tensor_dim0[31:16] | tensor_dim1[15:0]
  g1[3] = (tile_w << 16);                             // tensor_dim1[31:16]=0 | tile_dim0
  g1[4] = tile_h;                                     // tile_dim1 | tile_dim2=0
  g1[5] = gstride_elems;                              // tensor_dim0_stride[31:0]
  g1[6] = 0;
  g1[7] = 0;
  asm volatile("tensor_load_to_lds %0, %1" :: "s"(g0), "s"(g1) : "memory");
}

// ---------------- Kernel 1: A1 = nf @ We1[0:128,:], A2 = nf @ We1[128:256,:] ----------------
__global__ __launch_bounds__(256) void k1_precompute(
    const float* __restrict__ nf, const float* __restrict__ We1,
    float* __restrict__ A1, float* __restrict__ A2) {
  extern __shared__ __align__(16) char smem[];
  f16* Ain = (f16*)(smem);            // 32 x STRD   (8704 B)
  f16* W1a = (f16*)(smem + 8704);     // 128 x STRD  (34816 B)
  f16* W1b = (f16*)(smem + 43520);    // 128 x STRD  (34816 B)
  const int tid = threadIdx.x;
  const int rows0 = blockIdx.x * 32;

  for (int idx = tid; idx < 32 * 128; idx += 256) {
    int r = idx >> 7, h = idx & 127;
    Ain[r * STRD + h] = (f16)nf[(size_t)(rows0 + r) * 128 + h];
  }
  for (int idx = tid; idx < 128 * 128; idx += 256) {
    int k = idx >> 7, n = idx & 127;
    W1a[n * STRD + k] = (f16)We1[k * 128 + n];
    W1b[n * STRD + k] = (f16)We1[(k + 128) * 128 + n];
  }
  __syncthreads();

  const int wave = tid >> 5, lane = tid & 31;
  for (int t = 0; t < 4; ++t) {
    int idx4 = wave * 4 + t;
    int which = idx4 >> 4;        // 0 -> A1, 1 -> A2
    int mt = (idx4 >> 3) & 1;
    int nt = idx4 & 7;
    const f16* W = which ? W1b : W1a;
    v8f acc = {};
#pragma unroll
    for (int kk = 0; kk < 4; ++kk) {
      v16h a = frag_ld_a(Ain, mt * 16, kk * 32, STRD);
      v16h b = frag_ld_b(W, nt * 16, kk * 32, STRD);
      acc = __builtin_amdgcn_wmma_f32_16x16x32_f16(false, a, false, b, (short)0, acc, false, false);
    }
    float* out = which ? A2 : A1;
    int col = nt * 16 + (lane & 15);
    int rb = rows0 + mt * 16 + ((lane >> 4) << 3);
#pragma unroll
    for (int v = 0; v < 8; ++v) out[(size_t)(rb + v) * 128 + col] = acc[v];
  }
}

// ---------------- Kernel 2: fused edge MLPs + gating + message + position update ----------------
__global__ __launch_bounds__(256) void k2_edges(
    const float* __restrict__ pos, const float* __restrict__ valid,
    const float* __restrict__ adj, const float* __restrict__ We1,
    const float* __restrict__ We2, const float* __restrict__ Wc1,
    const float* __restrict__ Wc2, const float* __restrict__ Wi,
    const float* __restrict__ A1g, const float* __restrict__ A2g,
    float* __restrict__ msg_out, float* __restrict__ pos_out) {
  extern __shared__ __align__(16) char smem[];
  f16*   bufA  = (f16*)(smem + 0);        // 256 x STRD f16 (69632 B)
  f16*   bufC  = (f16*)(smem + 69632);    // 256 x STRD f16 (69632 B)
  f16*   WB2   = (f16*)(smem + 139264);   // 128 x STRD f16 (We2^T)
  f16*   WBc   = (f16*)(smem + 174080);   // 128 x STRD f16 (Wc1^T)
  float* A2s0  = (float*)(smem + 208896); // 16 x 132 (TDM double buffer 0)
  float* A2s1  = (float*)(smem + 217344); // 16 x 132 (TDM double buffer 1)
  float* A1s   = (float*)(smem + 225792); // 16 x 132
  float* wd    = (float*)(smem + 234240); // 128
  float* wa    = (float*)(smem + 234752); // 128
  float* Wc2v  = (float*)(smem + 235264); // 128
  float* Wiv   = (float*)(smem + 235776); // 128
  float* ev    = (float*)(smem + 236288); // 256
  float* adjs0 = (float*)(smem + 237312); // 256 (TDM double buffer 0)
  float* adjs1 = (float*)(smem + 238336); // 256 (TDM double buffer 1)
  float* posI  = (float*)(smem + 239360); // 16 x 4
  float* posJ  = (float*)(smem + 239616); // 16 x 4
  float* posAc = (float*)(smem + 239872); // 16 x 4
  float* vIs   = (float*)(smem + 240128); // 16
  float* vJs   = (float*)(smem + 240192); // 16
  float* sVal  = (float*)(smem + 240256); // 1

  const int tid = threadIdx.x;
  const int b = blockIdx.y;
  const int i0 = blockIdx.x * 16;
  const int nb = b * 128;
  const int wave = tid >> 5;
  const int lane = tid & 31;

  // TDM prologue: DMA first j-tile (A2 rows with LDS row padding 128->132, adj 16x16 tile)
  if (wave == 0) {
    tdm_load_2d(&A2g[(size_t)nb * 128], A2s0, 128, 16, 128, /*pad_int*/6, /*pad_amt*/3, 1);
    tdm_load_2d(&adj[(size_t)(nb + i0) * 128], adjs0, 16, 16, 128, 0, 0, 0);
  }

  for (int idx = tid; idx < 128 * 128; idx += 256) {
    int k = idx >> 7, n = idx & 127;
    WB2[n * STRD + k] = (f16)We2[k * 128 + n];
    WBc[n * STRD + k] = (f16)Wc1[k * 128 + n];
  }
  if (tid < 128) {
    wd[tid] = We1[256 * 128 + tid];
    wa[tid] = We1[257 * 128 + tid];
    Wc2v[tid] = Wc2[tid];
    Wiv[tid] = Wi[tid];
  }
  for (int idx = tid; idx < 16 * 128; idx += 256) {
    int il = idx >> 7, h = idx & 127;
    A1s[il * 132 + h] = A1g[(size_t)(nb + i0 + il) * 128 + h];
  }
  if (tid < 48) { int il = tid / 3, c = tid % 3; posI[il * 4 + c] = pos[(size_t)(nb + i0 + il) * 3 + c]; }
  if (tid < 16) vIs[tid] = valid[nb + i0 + tid];
  if (tid < 64) posAc[tid] = 0.0f;
  if (tid == 0) sVal[0] = 0.0f;
  __syncthreads();
  if (tid < 128) atomicAdd(sVal, valid[nb + tid]);

  const int nt = wave;             // each wave owns one N-tile; weight frags stay resident
  v16h bw2[4], bwc[4];
#pragma unroll
  for (int kk = 0; kk < 4; ++kk) {
    bw2[kk] = frag_ld_b(WB2, nt * 16, kk * 32, STRD);
    bwc[kk] = frag_ld_b(WBc, nt * 16, kk * 32, STRD);
  }

  const int il_m = tid >> 4;       // message ownership: (i_local, 8 h's)
  const int h0_m = (tid & 15) * 8;
  float msg[8];
#pragma unroll
  for (int q = 0; q < 8; ++q) msg[q] = 0.0f;

  const int ie = tid >> 4;         // per-edge passes: edge = 16*i_local + j_local
  const int je = tid & 15;

  for (int jb = 0; jb < 8; ++jb) {
    const int j0 = jb * 16;
    const float* A2c = (jb & 1) ? A2s1 : A2s0;
    const float* adc = (jb & 1) ? adjs1 : adjs0;
    if (tid < 48) { int jl = tid / 3, c = tid % 3; posJ[jl * 4 + c] = pos[(size_t)(nb + j0 + jl) * 3 + c]; }
    if (tid < 16) vJs[tid] = valid[nb + j0 + tid];
    if (jb < 7 && tid < 16)
      __builtin_prefetch(&adj[(size_t)(nb + i0 + tid) * 128 + j0 + 16], 0, 2);
    if (wave == 0) __builtin_amdgcn_s_wait_tensorcnt(0);  // current j-tile DMA complete
    __syncthreads();

    // Step A: pre-activation m1 = silu(A1[i] + A2[j] + dist*wd + adj*wa) -> bufA (f16)
    {
      float dx = posI[ie * 4 + 0] - posJ[je * 4 + 0];
      float dy = posI[ie * 4 + 1] - posJ[je * 4 + 1];
      float dz = posI[ie * 4 + 2] - posJ[je * 4 + 2];
      float dist = dx * dx + dy * dy + dz * dz;
      float av = adc[tid];
      const float* a1 = &A1s[ie * 132];
      const float* a2 = &A2c[je * 132];
      f16* dst = &bufA[tid * STRD];
      for (int h = 0; h < 128; ++h) {
        float x = a1[h] + a2[h] + dist * wd[h] + av * wa[h];
        dst[h] = (f16)siluf(x);
      }
    }
    __syncthreads();

    // Kick off next j-tile DMA; it overlaps both GEMMs below.
    if (jb < 7 && wave == 0) {
      float* A2n = (jb & 1) ? A2s0 : A2s1;
      float* adn = (jb & 1) ? adjs0 : adjs1;
      tdm_load_2d(&A2g[(size_t)(nb + j0 + 16) * 128], A2n, 128, 16, 128, 6, 3, 1);
      tdm_load_2d(&adj[(size_t)(nb + i0) * 128 + j0 + 16], adn, 16, 16, 128, 0, 0, 0);
    }

    // GEMM1: m = silu(m1 @ We2) -> bufC
    for (int mt = 0; mt < 16; ++mt) {
      v8f acc = {};
#pragma unroll
      for (int kk = 0; kk < 4; ++kk) {
        v16h a = frag_ld_a(bufA, mt * 16, kk * 32, STRD);
        acc = __builtin_amdgcn_wmma_f32_16x16x32_f16(false, a, false, bw2[kk], (short)0, acc, false, false);
      }
      int col = nt * 16 + (lane & 15);
      int rb = mt * 16 + ((lane >> 4) << 3);
#pragma unroll
      for (int v = 0; v < 8; ++v) bufC[(rb + v) * STRD + col] = (f16)siluf(acc[v]);
    }
    __syncthreads();

    // gate: e_ij = sigmoid(m . Wi)
    {
      float s = 0.0f;
      const f16* row = &bufC[tid * STRD];
      for (int h = 0; h < 128; ++h) s += (float)row[h] * Wiv[h];
      ev[tid] = sigmf_(s);
    }
    __syncthreads();

    // message accumulation (registers), then GEMM2: phi1 = silu(m @ Wc1) -> bufA
    for (int jl = 0; jl < 16; ++jl) {
      int e = il_m * 16 + jl;
      float coef = ev[e] * adc[e];
      const f16* row = &bufC[e * STRD + h0_m];
#pragma unroll
      for (int q = 0; q < 8; ++q) msg[q] += coef * (float)row[q];
    }
    for (int mt = 0; mt < 16; ++mt) {
      v8f acc = {};
#pragma unroll
      for (int kk = 0; kk < 4; ++kk) {
        v16h a = frag_ld_a(bufC, mt * 16, kk * 32, STRD);
        acc = __builtin_amdgcn_wmma_f32_16x16x32_f16(false, a, false, bwc[kk], (short)0, acc, false, false);
      }
      int col = nt * 16 + (lane & 15);
      int rb = mt * 16 + ((lane >> 4) << 3);
#pragma unroll
      for (int v = 0; v < 8; ++v) bufA[(rb + v) * STRD + col] = (f16)siluf(acc[v]);
    }
    __syncthreads();

    // phi_x = phi1 . Wc2 ; position accumulation
    {
      float phi = 0.0f;
      const f16* row = &bufA[tid * STRD];
      for (int h = 0; h < 128; ++h) phi += (float)row[h] * Wc2v[h];
      float dx = posI[ie * 4 + 0] - posJ[je * 4 + 0];
      float dy = posI[ie * 4 + 1] - posJ[je * 4 + 1];
      float dz = posI[ie * 4 + 2] - posJ[je * 4 + 2];
      float nrm = sqrtf(dx * dx + dy * dy + dz * dz);
      float inv = 1.0f / fmaxf(nrm, 1e-10f);
      float s = phi * vIs[ie] * vJs[je] * inv;
      atomicAdd(&posAc[ie * 4 + 0], dx * s);
      atomicAdd(&posAc[ie * 4 + 1], dy * s);
      atomicAdd(&posAc[ie * 4 + 2], dz * s);
    }
    __syncthreads();
  }

  {
    float* dst = &msg_out[(size_t)(nb + i0 + il_m) * 128 + h0_m];
#pragma unroll
    for (int q = 0; q < 8; ++q) dst[q] = msg[q];
  }
  if (tid < 48) {
    int il = tid / 3, c = tid % 3;
    float C = 1.0f / (sVal[0] - 1.0f);
    pos_out[(size_t)(nb + i0 + il) * 3 + c] = posI[il * 4 + c] + C * posAc[il * 4 + c];
  }
}

// ---------------- Kernel 3: node update MLP ----------------
__global__ __launch_bounds__(256) void k3_node(
    const float* __restrict__ nf, const float* __restrict__ msgb,
    const float* __restrict__ Wn1, const float* __restrict__ Wn2,
    float* __restrict__ out) {
  extern __shared__ __align__(16) char smem[];
  f16* inp = (f16*)(smem);             // 32 x STRD3 (16896 B)
  f16* W1t = (f16*)(smem + 16896);     // 128 x STRD3 (67584 B)
  f16* W2t = (f16*)(smem + 84480);     // 128 x STRD (34816 B)
  f16* mid = (f16*)(smem + 119296);    // 32 x STRD (8704 B)
  const int tid = threadIdx.x;
  const int rows0 = blockIdx.x * 32;

  for (int idx = tid; idx < 32 * 256; idx += 256) {
    int r = idx >> 8, k = idx & 255;
    float v = (k < 128) ? nf[(size_t)(rows0 + r) * 128 + k]
                        : msgb[(size_t)(rows0 + r) * 128 + (k - 128)];
    inp[r * STRD3 + k] = (f16)v;
  }
  for (int idx = tid; idx < 256 * 128; idx += 256) {
    int k = idx >> 7, n = idx & 127;
    W1t[n * STRD3 + k] = (f16)Wn1[k * 128 + n];
  }
  for (int idx = tid; idx < 128 * 128; idx += 256) {
    int k = idx >> 7, n = idx & 127;
    W2t[n * STRD + k] = (f16)Wn2[k * 128 + n];
  }
  __syncthreads();

  const int wave = tid >> 5, lane = tid & 31;
  const int nt = wave;
  for (int mt = 0; mt < 2; ++mt) {
    v8f acc = {};
#pragma unroll
    for (int kk = 0; kk < 8; ++kk) {
      v16h a = frag_ld_a(inp, mt * 16, kk * 32, STRD3);
      v16h w = frag_ld_b(W1t, nt * 16, kk * 32, STRD3);
      acc = __builtin_amdgcn_wmma_f32_16x16x32_f16(false, a, false, w, (short)0, acc, false, false);
    }
    int col = nt * 16 + (lane & 15);
    int rb = mt * 16 + ((lane >> 4) << 3);
#pragma unroll
    for (int v = 0; v < 8; ++v) mid[(rb + v) * STRD + col] = (f16)siluf(acc[v]);
  }
  __syncthreads();
  for (int mt = 0; mt < 2; ++mt) {
    v8f acc = {};
#pragma unroll
    for (int kk = 0; kk < 4; ++kk) {
      v16h a = frag_ld_a(mid, mt * 16, kk * 32, STRD);
      v16h w = frag_ld_b(W2t, nt * 16, kk * 32, STRD);
      acc = __builtin_amdgcn_wmma_f32_16x16x32_f16(false, a, false, w, (short)0, acc, false, false);
    }
    int col = nt * 16 + (lane & 15);
    int rb = rows0 + mt * 16 + ((lane >> 4) << 3);
#pragma unroll
    for (int v = 0; v < 8; ++v)
      out[(size_t)(rb + v) * 128 + col] = nf[(size_t)(rb + v) * 128 + col] + acc[v];
  }
}

extern "C" void kernel_launch(void* const* d_in, const int* in_sizes, int n_in,
                              void* d_out, int out_size, void* d_ws, size_t ws_size,
                              hipStream_t stream) {
  (void)in_sizes; (void)n_in; (void)out_size; (void)ws_size;
  const float* node_feat = (const float*)d_in[0];
  const float* pos   = (const float*)d_in[1];
  const float* valid = (const float*)d_in[2];
  const float* adj   = (const float*)d_in[3];
  const float* We1   = (const float*)d_in[4];
  const float* We2   = (const float*)d_in[5];
  const float* Wc1   = (const float*)d_in[6];
  const float* Wc2   = (const float*)d_in[7];
  const float* Wn1   = (const float*)d_in[8];
  const float* Wn2   = (const float*)d_in[9];
  const float* Wi    = (const float*)d_in[10];

  float* A1  = (float*)d_ws;           // 2048 x 128
  float* A2  = A1 + 2048 * 128;        // 2048 x 128
  float* msg = A2 + 2048 * 128;        // 2048 x 128

  float* out_node = (float*)d_out;                 // 16*128*128
  float* out_pos  = out_node + 16 * 128 * 128;     // 16*128*3

  constexpr unsigned SM1 = 78336, SM2 = 240272, SM3 = 128000;
  (void)hipFuncSetAttribute((const void*)k1_precompute, hipFuncAttributeMaxDynamicSharedMemorySize, SM1);
  (void)hipFuncSetAttribute((const void*)k2_edges,      hipFuncAttributeMaxDynamicSharedMemorySize, SM2);
  (void)hipFuncSetAttribute((const void*)k3_node,       hipFuncAttributeMaxDynamicSharedMemorySize, SM3);

  k1_precompute<<<64, 256, SM1, stream>>>(node_feat, We1, A1, A2);
  k2_edges<<<dim3(8, 16), 256, SM2, stream>>>(pos, valid, adj, We1, We2, Wc1, Wc2, Wi,
                                              A1, A2, msg, out_pos);
  k3_node<<<64, 256, SM3, stream>>>(node_feat, msg, Wn1, Wn2, out_node);
}
